// Decoder_6708738916935
// MI455X (gfx1250) — compile-verified, matrix operand order checked by reading, not run on hardware
//
#include <hip/hip_runtime.h>
#include <hip/hip_bf16.h>

// ---------------- problem constants ----------------
#define BB   512            // batch
#define TT   128            // tgt len
#define SS   128            // enc len
#define EE   256            // emb dim
#define ENCD 1024           // enc feature
#define HH   512            // dec hidden
#define VV   128            // vocab
#define XK   (EE + ENCD + HH)   // 1792 : fused operand [e | ctx | h]
#define G4H  (4 * HH)           // 2048

typedef unsigned short u16;
typedef __attribute__((ext_vector_type(16))) __bf16 v16bf;
typedef __attribute__((ext_vector_type(8)))  float  v8f;

struct __align__(16) B128 { unsigned int w[4]; };
union V16U { v16bf v; B128 q[2]; };

__device__ __forceinline__ u16 f2bf(float f) {
    unsigned u = __float_as_uint(f);
    unsigned r = u + 0x7FFFu + ((u >> 16) & 1u);   // round-to-nearest-even
    return (u16)(r >> 16);
}
__device__ __forceinline__ float bf2f(u16 h) {
    return __uint_as_float(((unsigned)h) << 16);
}

// Load 16 bf16 per lane in the CDNA5 16-bit A/B operand layout:
// halves [0..7]  = K = kbase+0..7,  halves [8..15] = K = kbase+16..23
__device__ __forceinline__ v16bf load_op16(const u16* row, int kbase) {
    V16U t;
    t.q[0] = *reinterpret_cast<const B128*>(row + kbase);
    t.q[1] = *reinterpret_cast<const B128*>(row + kbase + 16);
    return t.v;
}

// ---------------- WMMA GEMM: C[M,N] = A[M,K] * B[N,K]^T (+bias[n]) ----------
// A: bf16 row-major, leading dim lda; B: bf16 row-major [N,K], leading dim ldb.
// Register-blocked: each wave computes a 32x32 tile (2x2 WMMA subtiles);
// 4 waves/block span 32 rows x 128 cols. Requires M % 32 == 0, N % 128 == 0.
__global__ __launch_bounds__(128)
void k_gemm_bf16(const u16* __restrict__ A, int lda,
                 const u16* __restrict__ B, int ldb,
                 float* __restrict__ C, int ldc,
                 int N, int K, const float* __restrict__ bias) {
    const int lane = threadIdx.x & 31;
    const int wave = threadIdx.x >> 5;
    const int n0 = blockIdx.x * 128 + wave * 32;
    const int m0 = blockIdx.y * 32;

    const int r    = lane & 15;
    const int kOff = (lane < 16) ? 0 : 8;      // lanes 16-31 hold K+8..15 / K+24..31
    const u16* Arow0 = A + (size_t)(m0 + r)      * lda;
    const u16* Arow1 = A + (size_t)(m0 + 16 + r) * lda;
    const u16* Brow0 = B + (size_t)(n0 + r)      * ldb;
    const u16* Brow1 = B + (size_t)(n0 + 16 + r) * ldb;

    v8f acc00 = {}, acc01 = {}, acc10 = {}, acc11 = {};
    for (int k0 = 0; k0 < K; k0 += 32) {
        const int kb = k0 + kOff;
        v16bf a0 = load_op16(Arow0, kb);
        v16bf a1 = load_op16(Arow1, kb);
        v16bf b0 = load_op16(Brow0, kb);
        v16bf b1 = load_op16(Brow1, kb);
        acc00 = __builtin_amdgcn_wmma_f32_16x16x32_bf16(false, a0, false, b0,
                                                        (short)0, acc00, false, false);
        acc01 = __builtin_amdgcn_wmma_f32_16x16x32_bf16(false, a0, false, b1,
                                                        (short)0, acc01, false, false);
        acc10 = __builtin_amdgcn_wmma_f32_16x16x32_bf16(false, a1, false, b0,
                                                        (short)0, acc10, false, false);
        acc11 = __builtin_amdgcn_wmma_f32_16x16x32_bf16(false, a1, false, b1,
                                                        (short)0, acc11, false, false);
    }

    // C/D layout: VGPR i, lane l -> row = 8*(l>>4) + i, col = l & 15
    const int mBase = m0 + (lane >> 4) * 8;
    const int nA = n0 + (lane & 15);
    const int nB = nA + 16;
    const float bvA = bias ? bias[nA] : 0.0f;
    const float bvB = bias ? bias[nB] : 0.0f;
#pragma unroll
    for (int i = 0; i < 8; ++i) {
        C[(size_t)(mBase + i)      * ldc + nA] = acc00[i] + bvA;
        C[(size_t)(mBase + i)      * ldc + nB] = acc01[i] + bvB;
        C[(size_t)(mBase + 16 + i) * ldc + nA] = acc10[i] + bvA;
        C[(size_t)(mBase + 16 + i) * ldc + nB] = acc11[i] + bvB;
    }
}

// ---------------- one-time setup kernels ----------------
__global__ void k_cvt_bf16(const float* __restrict__ s, u16* __restrict__ d, int n) {
    int i = blockIdx.x * blockDim.x + threadIdx.x;
    if (i < n) d[i] = f2bf(s[i]);
}
__global__ void k_zero_u16(u16* p, int n) {
    int i = blockIdx.x * blockDim.x + threadIdx.x;
    if (i < n) p[i] = 0;
}
__global__ void k_zero_f32(float* p, int n) {
    int i = blockIdx.x * blockDim.x + threadIdx.x;
    if (i < n) p[i] = 0.0f;
}
// out[:,0,:] = 0
__global__ void k_zero_out0(float* __restrict__ out) {
    int i = blockIdx.x * blockDim.x + threadIdx.x;   // over B*V
    if (i < BB * VV) {
        int b = i / VV, v = i % VV;
        out[(size_t)b * TT * VV + v] = 0.0f;
    }
}
// emb -> bf16, row PAD_IDX(0) zeroed
__global__ void k_build_emb(const float* __restrict__ emb, u16* __restrict__ d) {
    int i = blockIdx.x * blockDim.x + threadIdx.x;   // V*E
    if (i < VV * EE) {
        int row = i / EE;
        d[i] = (row == 0) ? (u16)0 : f2bf(emb[i]);
    }
}
// Wcat[n, 0:1280]=W_ih[n,:], Wcat[n, 1280:1792]=W_hh[n,:]   (matches xh=[e|ctx|h])
__global__ void k_build_wcat(const float* __restrict__ wih, const float* __restrict__ whh,
                             u16* __restrict__ d) {
    int i = blockIdx.x * blockDim.x + threadIdx.x;   // 4H * XK
    if (i < G4H * XK) {
        int n = i / XK, c = i % XK;
        float v = (c < EE + ENCD) ? wih[(size_t)n * (EE + ENCD) + c]
                                  : whh[(size_t)n * HH + (c - (EE + ENCD))];
        d[i] = f2bf(v);
    }
}
// W_out columns are [h(512) | ctx(1024) | e(256)]; reorder to xh's [e|ctx|h]
__global__ void k_build_woutr(const float* __restrict__ wout, u16* __restrict__ d) {
    int i = blockIdx.x * blockDim.x + threadIdx.x;   // V * XK
    if (i < VV * XK) {
        int v = i / XK, c = i % XK;
        int src;
        if (c < EE)              src = HH + ENCD + c;        // e part
        else if (c < EE + ENCD)  src = HH + (c - EE);        // ctx part
        else                     src = c - (EE + ENCD);      // h part
        d[i] = f2bf(wout[(size_t)v * XK + src]);
    }
}
__global__ void k_bias_sum(const float* __restrict__ a, const float* __restrict__ b,
                           float* __restrict__ d) {
    int i = blockIdx.x * blockDim.x + threadIdx.x;
    if (i < G4H) d[i] = a[i] + b[i];
}

// ---------------- per-step kernels ----------------
// gather this step's embeddings into xh[:, 0:E]
__global__ void k_embed_step(const int* __restrict__ tgt, const u16* __restrict__ embbf,
                             u16* __restrict__ xh, int t) {
    int i = blockIdx.x * blockDim.x + threadIdx.x;   // B*E
    if (i < BB * EE) {
        int b = i / EE, e = i % EE;
        int tok = tgt[(size_t)b * TT + t];
        xh[(size_t)b * XK + e] = embbf[(size_t)tok * EE + e];
    }
}

// attention: one block (256 thr) per batch. scores -> masked softmax -> ctx(bf16)
__global__ __launch_bounds__(256)
void k_attn(const u16* __restrict__ encbf, const int* __restrict__ enc_lens,
            const float* __restrict__ proj, u16* __restrict__ xh) {
    __shared__ float projs[ENCD];
    __shared__ float sc[SS];
    __shared__ float red[8];

    const int b    = blockIdx.x;
    const int tid  = threadIdx.x;
    const int lane = tid & 31;
    const int wave = tid >> 5;
    const u16* encb = encbf + (size_t)b * SS * ENCD;

    for (int e = tid; e < ENCD; e += 256) projs[e] = proj[(size_t)b * ENCD + e];
    __syncthreads();

    // scores[s] = dot(enc[b,s,:], proj[b,:])   (8 waves x 16 rows each)
    for (int s = wave; s < SS; s += 8) {
        const u16* row = encb + (size_t)s * ENCD;
        float sum = 0.0f;
        for (int e2 = lane; e2 < ENCD / 2; e2 += 32) {
            unsigned pk = *reinterpret_cast<const unsigned*>(row + 2 * e2);
            sum += bf2f((u16)(pk & 0xFFFF)) * projs[2 * e2];
            sum += bf2f((u16)(pk >> 16))    * projs[2 * e2 + 1];
        }
#pragma unroll
        for (int o = 16; o > 0; o >>= 1) sum += __shfl_xor(sum, o, 32);
        if (lane == 0) sc[s] = sum;
    }
    __syncthreads();

    const int len = enc_lens[b];
    // masked max
    if (tid < SS) {
        float x = (tid < len) ? sc[tid] : -1e9f;
        float m = x;
#pragma unroll
        for (int o = 16; o > 0; o >>= 1) m = fmaxf(m, __shfl_xor(m, o, 32));
        if (lane == 0) red[wave] = m;
    }
    __syncthreads();
    if (tid == 0) red[0] = fmaxf(fmaxf(red[0], red[1]), fmaxf(red[2], red[3]));
    __syncthreads();
    const float gm = red[0];
    // exp + sum
    if (tid < SS) {
        float x = (tid < len) ? sc[tid] : -1e9f;
        float ex = __expf(x - gm);
        sc[tid] = ex;
        float s = ex;
#pragma unroll
        for (int o = 16; o > 0; o >>= 1) s += __shfl_xor(s, o, 32);
        if (lane == 0) red[4 + wave] = s;
    }
    __syncthreads();
    if (tid == 0) red[4] = red[4] + red[5] + red[6] + red[7];
    __syncthreads();
    const float inv = 1.0f / red[4];
    if (tid < SS) sc[tid] *= inv;
    __syncthreads();

    // ctx[e] = sum_s aw[s] * enc[b,s,e]  -> xh[:, E : E+ENC] (bf16)
    for (int e = tid; e < ENCD; e += 256) {
        float acc = 0.0f;
        for (int s = 0; s < SS; ++s)
            acc += sc[s] * bf2f(encb[(size_t)s * ENCD + e]);
        xh[(size_t)b * XK + EE + e] = f2bf(acc);
    }
}

// LSTM cell: consumes gates[B,4H] (order i,f,g,o), updates c, writes h(bf16) into xh
__global__ void k_cell(const float* __restrict__ gates, float* __restrict__ c,
                       u16* __restrict__ xh) {
    int idx = blockIdx.x * blockDim.x + threadIdx.x;   // B*H
    if (idx >= BB * HH) return;
    int b = idx >> 9, j = idx & (HH - 1);
    const float* g = gates + (size_t)b * G4H;
    float ig = g[j], fg = g[HH + j], gg = g[2 * HH + j], og = g[3 * HH + j];
    float si = 1.0f / (1.0f + __expf(-ig));
    float sf = 1.0f / (1.0f + __expf(-fg));
    float so = 1.0f / (1.0f + __expf(-og));
    float cn = sf * c[idx] + si * tanhf(gg);
    c[idx] = cn;
    xh[(size_t)b * XK + EE + ENCD + j] = f2bf(so * tanhf(cn));
}

// ---------------- launch ----------------
extern "C" void kernel_launch(void* const* d_in, const int* in_sizes, int n_in,
                              void* d_out, int out_size, void* d_ws, size_t ws_size,
                              hipStream_t stream) {
    const int*   tgt      = (const int*)  d_in[0];
    const float* enc      = (const float*)d_in[1];
    const int*   enc_lens = (const int*)  d_in[2];
    const float* emb      = (const float*)d_in[3];
    const float* W_ih     = (const float*)d_in[4];
    const float* W_hh     = (const float*)d_in[5];
    const float* b_ih     = (const float*)d_in[6];
    const float* b_hh     = (const float*)d_in[7];
    const float* W_attn   = (const float*)d_in[8];
    const float* W_out    = (const float*)d_in[9];
    const float* b_out    = (const float*)d_in[10];
    float* out = (float*)d_out;

    // ---- workspace carve-up (all sizes multiples of 256 B) ----
    char* ws = (char*)d_ws;
    size_t off = 0;
    u16*   enc_bf  = (u16*)(ws + off); off += (size_t)BB * SS * ENCD * 2;  // 128 MB
    u16*   wcat    = (u16*)(ws + off); off += (size_t)G4H * XK * 2;        // 7.0 MB
    u16*   woutr   = (u16*)(ws + off); off += (size_t)VV * XK * 2;
    u16*   wattnbf = (u16*)(ws + off); off += (size_t)ENCD * HH * 2;
    u16*   embbf   = (u16*)(ws + off); off += (size_t)VV * EE * 2;
    float* biasg   = (float*)(ws + off); off += (size_t)G4H * 4;
    u16*   xh      = (u16*)(ws + off); off += (size_t)BB * XK * 2;         // [e|ctx|h]
    float* proj    = (float*)(ws + off); off += (size_t)BB * ENCD * 4;
    float* gates   = (float*)(ws + off); off += (size_t)BB * G4H * 4;
    float* cbuf    = (float*)(ws + off); off += (size_t)BB * HH * 4;
    (void)ws_size; (void)in_sizes; (void)n_in; (void)out_size;

    const int TPB = 256;
    #define NB(n) (((n) + TPB - 1) / TPB)

    // ---- one-time setup ----
    k_cvt_bf16  <<<NB(BB * SS * ENCD), TPB, 0, stream>>>(enc, enc_bf, BB * SS * ENCD);
    k_build_wcat<<<NB(G4H * XK),       TPB, 0, stream>>>(W_ih, W_hh, wcat);
    k_build_woutr<<<NB(VV * XK),       TPB, 0, stream>>>(W_out, woutr);
    k_cvt_bf16  <<<NB(ENCD * HH),      TPB, 0, stream>>>(W_attn, wattnbf, ENCD * HH);
    k_build_emb <<<NB(VV * EE),        TPB, 0, stream>>>(emb, embbf);
    k_bias_sum  <<<NB(G4H),            TPB, 0, stream>>>(b_ih, b_hh, biasg);
    k_zero_u16  <<<NB(BB * XK),        TPB, 0, stream>>>(xh, BB * XK);      // h0 = 0
    k_zero_f32  <<<NB(BB * HH),        TPB, 0, stream>>>(cbuf, BB * HH);    // c0 = 0
    k_zero_out0 <<<NB(BB * VV),        TPB, 0, stream>>>(out);              // out[:,0,:]=0

    // ---- 127 teacher-forced steps ----
    for (int t = 0; t < TT - 1; ++t) {
        // e_t -> xh[:, 0:E]
        k_embed_step<<<NB(BB * EE), TPB, 0, stream>>>(tgt, embbf, xh, t);

        // proj[B,ENC] = h * W_attn^T    (A = xh h-slot, K = H)
        k_gemm_bf16<<<dim3(ENCD / 128, BB / 32), 128, 0, stream>>>(
            xh + (EE + ENCD), XK, wattnbf, HH, proj, ENCD, ENCD, HH, nullptr);

        // attention -> ctx in xh[:, E:E+ENC]
        k_attn<<<BB, 256, 0, stream>>>(enc_bf, enc_lens, proj, xh);

        // gates[B,4H] = [e|ctx|h] * Wcat^T + (b_ih+b_hh)
        k_gemm_bf16<<<dim3(G4H / 128, BB / 32), 128, 0, stream>>>(
            xh, XK, wcat, XK, gates, G4H, G4H, XK, biasg);

        // LSTM cell -> c, h(bf16) in xh h-slot
        k_cell<<<NB(BB * HH), TPB, 0, stream>>>(gates, cbuf, xh);

        // logits = [e|ctx|h] * Wout_r^T + b_out  -> out[:, t+1, :]
        k_gemm_bf16<<<dim3(VV / 128, BB / 32), 128, 0, stream>>>(
            xh, XK, woutr, XK, out + (size_t)(t + 1) * VV, TT * VV, VV, XK, b_out);
    }
    #undef NB
}
